// BiLSTM_CRF_63754494542060
// MI455X (gfx1250) — matrix operand order repaired and checked by reading, not compiled, not run on hardware
//
#include <hip/hip_runtime.h>
#include <hip/hip_bf16.h>

// ---------------------------------------------------------------------------
// BiLSTM-CRF for MI455X (gfx1250): bf16 WMMA GEMMs with async-to-LDS staging
// for the large input projection, per-step recurrence, LDS-resident CRF.
// B=64, T=512, E=1024, H=512, 4H=2048, K=74 (padded to 80 for WMMA N tiles)
// ---------------------------------------------------------------------------

typedef __attribute__((ext_vector_type(16))) __bf16 v16bf;
typedef __attribute__((ext_vector_type(8)))  float  v8f;

#define BB   64
#define TT   512
#define EE   1024
#define HH   512
#define G4H  2048
#define KK   74
#define KPAD 80

// fp32 -> bf16 bits, round-to-nearest-even-ish
static __device__ __forceinline__ unsigned short f2bf_bits(float f) {
    union { float f; unsigned u; } a; a.f = f;
    unsigned r = a.u + 0x7FFFu + ((a.u >> 16) & 1u);
    return (unsigned short)(r >> 16);
}

// --- WMMA fragment loaders (ISA 7.12.2 layouts, wave32) ---------------------
// A (16x32 bf16): lane l -> row = base+(l&15); kb=(l>=16)?8:0;
//                 values K = kb..kb+7 then kb+16..kb+23 (contiguous 16B chunks)
static __device__ __forceinline__ v16bf frag_A(const unsigned short* P, int ld,
                                               int row, int k0, int lane) {
    int half = lane >> 4;
    const unsigned short* p = P + (size_t)row * ld + k0 + half * 8;
    union { uint4 q[2]; v16bf v; } u;
    u.q[0] = *(const uint4*)(p);
    u.q[1] = *(const uint4*)(p + 16);
    return u.v;
}
// B (32x16 bf16): lane l -> col = base+(l&15); kb=(l>=16)?8:0; K = kb..kb+7 then kb+16..kb+23
static __device__ __forceinline__ v16bf frag_B(const unsigned short* P, int ld,
                                               int col, int k0, int lane) {
    int half = lane >> 4;
    const unsigned short* p = P + (size_t)col * ld + k0 + half * 8;
    union { uint4 q[2]; v16bf v; } u;
    u.q[0] = *(const uint4*)(p);
    u.q[1] = *(const uint4*)(p + 16);
    return u.v;
}

// --- async global->LDS copy (gfx1250, ASYNCcnt path) ------------------------
static __device__ __forceinline__ void async_copy16(unsigned lds_off,
                                                    const unsigned short* gptr) {
    unsigned long long ga = (unsigned long long)(uintptr_t)gptr;
    asm volatile("global_load_async_to_lds_b128 %0, %1, off"
                 :: "v"(lds_off), "v"(ga) : "memory");
}
static __device__ __forceinline__ void wait_async0() {
    asm volatile("s_wait_asynccnt 0x0" ::: "memory");
}
// stage a 64x32 bf16 tile (4KB) from src rows [row0,row0+64), k [k0,k0+32).
// 256 16B chunks / 128 threads -> exactly 2 unconditional issues per thread
// (straight-line: no exec masking in the hot loop).
static __device__ __forceinline__ void stage_tile(unsigned ldsbase,
                                                  const unsigned short* src,
                                                  int ld, int row0, int k0, int tid) {
    {
        int c = tid;                              // chunk 0..127
        int row = c >> 2, koff = (c & 3) * 8;
        async_copy16(ldsbase + (unsigned)(row * 32 + koff) * 2,
                     src + (size_t)(row0 + row) * ld + k0 + koff);
    }
    {
        int c = tid + 128;                        // chunk 128..255
        int row = c >> 2, koff = (c & 3) * 8;
        async_copy16(ldsbase + (unsigned)(row * 32 + koff) * 2,
                     src + (size_t)(row0 + row) * ld + k0 + koff);
    }
}

// --- small conversion / packing kernels -------------------------------------
__global__ void k_f2bf(const float* __restrict__ src,
                       unsigned short* __restrict__ dst, int n) {
    int i = blockIdx.x * blockDim.x + threadIdx.x;
    if (i < n) dst[i] = f2bf_bits(src[i]);
}

__global__ void k_packwout(const float* __restrict__ w,
                           unsigned short* __restrict__ dst) {
    int i = blockIdx.x * blockDim.x + threadIdx.x;   // KPAD*EE elements
    if (i >= KPAD * (EE)) return;
    int n = i >> 10, e = i & 1023;
    float v = (n < KK) ? w[(size_t)n * (EE) + e] : 0.f;
    dst[i] = f2bf_bits(v);
}

__global__ void k_bias(const float* __restrict__ a, const float* __restrict__ b,
                       float* __restrict__ o) {
    int i = blockIdx.x * blockDim.x + threadIdx.x;
    if (i < G4H) o[i] = a[i] + b[i];
}

__global__ void k_init(const float* __restrict__ h0, const float* __restrict__ c0,
                       unsigned short* __restrict__ hbf, float* __restrict__ c) {
    int i = blockIdx.x * blockDim.x + threadIdx.x;   // B*H
    if (i < BB * HH) { hbf[i] = f2bf_bits(h0[i]); c[i] = c0[i]; }
}

// --- input-projection GEMM: PG[t*B+b][n] = X[b*T+t] . Wih[n] + bias[n] ------
// M=B*T=32768 (row m=b*T+t), N=4H=2048, K=E=1024.
// 64x64 block tile, 4 waves (each 16Mx64N = 4 accumulators), double-buffered
// async-to-LDS staging of A/B k-panels.
__global__ __launch_bounds__(128) void k_pregates(
    const unsigned short* __restrict__ Xbf, const unsigned short* __restrict__ W,
    const float* __restrict__ bias, float* __restrict__ PG) {
    __shared__ unsigned short sA[2][64 * 32];
    __shared__ unsigned short sB[2][64 * 32];
    int tid = threadIdx.x, lane = tid & 31, wave = tid >> 5;
    int r = lane & 15, half = lane >> 4;
    int tileM = blockIdx.y * 64, tileN = blockIdx.x * 64;

    v8f acc[4];
#pragma unroll
    for (int nt = 0; nt < 4; nt++) {
        float bv = bias[tileN + nt * 16 + r];
#pragma unroll
        for (int j = 0; j < 8; j++) acc[nt][j] = bv;
    }

    unsigned aB[2] = { (unsigned)(uintptr_t)&sA[0][0], (unsigned)(uintptr_t)&sA[1][0] };
    unsigned bB[2] = { (unsigned)(uintptr_t)&sB[0][0], (unsigned)(uintptr_t)&sB[1][0] };

    // prologue: stage k-chunk 0 into buffer 0
    stage_tile(aB[0], Xbf, EE, tileM, 0, tid);
    stage_tile(bB[0], W,   EE, tileN, 0, tid);

    for (int k0 = 0; k0 < EE; k0 += 32) {
        int cur = (k0 >> 5) & 1, nxt = cur ^ 1;
        wait_async0();          // own staged chunk complete
        __syncthreads();        // everyone's chunk complete; prev reads retired
        if (k0 + 32 < EE) {     // overlap next stage with current compute
            stage_tile(aB[nxt], Xbf, EE, tileM, k0 + 32, tid);
            stage_tile(bB[nxt], W,   EE, tileN, k0 + 32, tid);
        }
        v16bf a = frag_A(&sA[cur][0], 32, wave * 16 + r, 0, lane);
#pragma unroll
        for (int nt = 0; nt < 4; nt++) {
            v16bf b = frag_B(&sB[cur][0], 32, nt * 16 + r, 0, lane);
            acc[nt] = __builtin_amdgcn_wmma_f32_16x16x32_bf16(false, a, false, b,
                                                              (short)0, acc[nt],
                                                              false, false);
        }
    }

#pragma unroll
    for (int nt = 0; nt < 4; nt++) {
        int n = tileN + nt * 16 + r;
#pragma unroll
        for (int j = 0; j < 8; j++) {
            int m = tileM + wave * 16 + j + half * 8;  // m = b*T + t
            int bb = m >> 9, t = m & 511;
            PG[((size_t)(t * BB + bb)) * G4H + n] = acc[nt][j];
        }
    }
}

// --- recurrent GEMM: gtmp[b][n] = PGt[b][n] + h[b] . Whh[n] -----------------
// M=B=64, N=4H=2048, K=H=512; per-wave 16x16 tiles (max step parallelism)
__global__ __launch_bounds__(128) void k_recgemm(
    const unsigned short* __restrict__ Hbf, const unsigned short* __restrict__ Whh,
    const float* __restrict__ PGt, float* __restrict__ gtmp) {
    int lane = threadIdx.x & 31, wave = threadIdx.x >> 5;
    int tileN = (blockIdx.x * 4 + wave) * 16;
    int tileM = blockIdx.y * 16;
    int r = lane & 15, half = lane >> 4;
    int rowA = tileM + r, colB = tileN + r;
    v8f acc;
#pragma unroll
    for (int j = 0; j < 8; j++)
        acc[j] = PGt[(size_t)(tileM + j + half * 8) * G4H + colB];
    for (int k0 = 0; k0 < HH; k0 += 32) {
        v16bf a = frag_A(Hbf, HH, rowA, k0, lane);
        v16bf b = frag_B(Whh, HH, colB, k0, lane);
        acc = __builtin_amdgcn_wmma_f32_16x16x32_bf16(false, a, false, b,
                                                      (short)0, acc, false, false);
    }
#pragma unroll
    for (int j = 0; j < 8; j++)
        gtmp[(size_t)(tileM + j + half * 8) * G4H + colB] = acc[j];
}

// --- LSTM cell elementwise update (torch gate order i,f,g,o) ----------------
__global__ void k_cell(const float* __restrict__ g, float* __restrict__ c,
                       unsigned short* __restrict__ hbf_cur,
                       unsigned short* __restrict__ hbf_t) {
    int idx = blockIdx.x * blockDim.x + threadIdx.x;  // B*H = 32768
    if (idx >= BB * HH) return;
    int b = idx >> 9, j = idx & 511;
    const float* gb = g + (size_t)b * G4H;
    float gi = gb[j], gf = gb[HH + j], gg = gb[2 * HH + j], go = gb[3 * HH + j];
    float si = 1.f / (1.f + __expf(-gi));
    float sf = 1.f / (1.f + __expf(-gf));
    float so = 1.f / (1.f + __expf(-go));
    float cc = sf * c[idx] + si * tanhf(gg);
    float h  = so * tanhf(cc);
    c[idx] = cc;
    unsigned short hb = f2bf_bits(h);
    hbf_cur[idx] = hb;
    hbf_t[idx]   = hb;
}

// --- output projection: em[m][k] = concat(hf[m],hb[m]) . Wout[k] + bout[k] --
// M=T*B=32768 (row m=t*B+b), N=KPAD=80 (store k<74), K=HID=1024
__global__ __launch_bounds__(32) void k_outproj(
    const unsigned short* __restrict__ hf, const unsigned short* __restrict__ hb,
    const unsigned short* __restrict__ Wout, const float* __restrict__ bout,
    float* __restrict__ em) {
    int lane = threadIdx.x & 31;
    int tileN = blockIdx.x * 16;
    int tileM = blockIdx.y * 16;
    int r = lane & 15, half = lane >> 4;
    int rowA = tileM + r, colB = tileN + r;
    v8f acc;
#pragma unroll
    for (int j = 0; j < 8; j++) acc[j] = 0.f;
    for (int k0 = 0; k0 < EE; k0 += 32) {   // concat K = 1024 (hf|hb)
        union { uint4 q[2]; v16bf v; } ua;
        int ka = k0 + half * 8;
        {
            int k = ka;
            const unsigned short* p = (k < HH) ? hf + (size_t)rowA * HH + k
                                               : hb + (size_t)rowA * HH + (k - HH);
            ua.q[0] = *(const uint4*)p;
        }
        {
            int k = ka + 16;
            const unsigned short* p = (k < HH) ? hf + (size_t)rowA * HH + k
                                               : hb + (size_t)rowA * HH + (k - HH);
            ua.q[1] = *(const uint4*)p;
        }
        v16bf b = frag_B(Wout, EE, colB, k0, lane);
        acc = __builtin_amdgcn_wmma_f32_16x16x32_bf16(false, ua.v, false, b,
                                                      (short)0, acc, false, false);
    }
    if (colB < KK) {
        float bo = bout[colB];
#pragma unroll
        for (int j = 0; j < 8; j++)
            em[(size_t)(tileM + j + half * 8) * KK + colB] = acc[j] + bo;
    }
}

// --- CRF: one workgroup per sequence; alpha + trans in LDS ------------------
__global__ __launch_bounds__(128) void k_crf(
    const float* __restrict__ em, const int* __restrict__ tags,
    const int* __restrict__ lengths, const float* __restrict__ start,
    const float* __restrict__ endt, const float* __restrict__ trans,
    float* __restrict__ nll) {
    __shared__ float s_tr[KK * KK];
    __shared__ float s_al[KK];
    int b = blockIdx.x, tid = threadIdx.x;
    for (int i = tid; i < KK * KK; i += 128) s_tr[i] = trans[i];
    int len = lengths[b];
    if (tid < KK) s_al[tid] = start[tid] + em[(size_t)b * KK + tid];
    __syncthreads();
    for (int t = 1; t < len; t++) {
        const float* emt = em + ((size_t)t * BB + b) * KK;
        float v = 0.f;
        if (tid < KK) {
            float mx = -3.4e38f;
            for (int i = 0; i < KK; i++)
                mx = fmaxf(mx, s_al[i] + s_tr[i * KK + tid]);
            float s = 0.f;
            for (int i = 0; i < KK; i++)
                s += __expf(s_al[i] + s_tr[i * KK + tid] - mx);
            v = mx + __logf(s) + emt[tid];
        }
        __syncthreads();
        if (tid < KK) s_al[tid] = v;
        __syncthreads();
    }
    if (tid == 0) {
        float mx = -3.4e38f;
        for (int j = 0; j < KK; j++) mx = fmaxf(mx, s_al[j] + endt[j]);
        float s = 0.f;
        for (int j = 0; j < KK; j++) s += __expf(s_al[j] + endt[j] - mx);
        float logZ = mx + __logf(s);
        const int* tg = tags + (size_t)b * TT;
        int prev = tg[0];
        float score = start[prev] + em[(size_t)b * KK + prev];
        for (int t = 1; t < len; t++) {
            int cur = tg[t];
            score += s_tr[prev * KK + cur] + em[((size_t)t * BB + b) * KK + cur];
            prev = cur;
        }
        score += endt[prev];
        nll[b] = logZ - score;
    }
}

__global__ void k_reduce(const float* __restrict__ nll, float* __restrict__ out) {
    __shared__ float s[64];
    int tid = threadIdx.x;
    s[tid] = nll[tid];
    __syncthreads();
    for (int st = 32; st > 0; st >>= 1) {
        if (tid < st) s[tid] += s[tid + st];
        __syncthreads();
    }
    if (tid == 0) out[0] = s[0] / 64.f;
}

// ---------------------------------------------------------------------------
extern "C" void kernel_launch(void* const* d_in, const int* in_sizes, int n_in,
                              void* d_out, int out_size, void* d_ws, size_t ws_size,
                              hipStream_t stream) {
    const float* embeds  = (const float*)d_in[0];
    const int*   tag_ids = (const int*)d_in[1];
    const int*   lengths = (const int*)d_in[2];
    const float* h0      = (const float*)d_in[3];
    const float* c0      = (const float*)d_in[4];
    const float* w_ih_f  = (const float*)d_in[5];
    const float* w_hh_f  = (const float*)d_in[6];
    const float* b_ih_f  = (const float*)d_in[7];
    const float* b_hh_f  = (const float*)d_in[8];
    const float* w_ih_b  = (const float*)d_in[9];
    const float* w_hh_b  = (const float*)d_in[10];
    const float* b_ih_b  = (const float*)d_in[11];
    const float* b_hh_b  = (const float*)d_in[12];
    const float* w_out   = (const float*)d_in[13];
    const float* b_out   = (const float*)d_in[14];
    const float* start_t = (const float*)d_in[15];
    const float* end_t   = (const float*)d_in[16];
    const float* trans   = (const float*)d_in[17];

    // workspace carve-up (256B aligned)
    char* base = (char*)d_ws;
    size_t off = 0;
    auto alloc = [&](size_t bytes) -> char* {
        char* p = base + off;
        off += (bytes + 255) & ~(size_t)255;
        return p;
    };
    float*          PG    = (float*)alloc((size_t)TT * BB * G4H * 4);   // 256 MB
    unsigned short* Xbf   = (unsigned short*)alloc((size_t)BB * TT * EE * 2);
    unsigned short* HFbf  = (unsigned short*)alloc((size_t)TT * BB * HH * 2);
    unsigned short* HBbf  = (unsigned short*)alloc((size_t)TT * BB * HH * 2);
    unsigned short* Hcur  = (unsigned short*)alloc((size_t)BB * HH * 2);
    float*          Ccur  = (float*)alloc((size_t)BB * HH * 4);
    float*          Gtmp  = (float*)alloc((size_t)BB * G4H * 4);
    unsigned short* Wihf  = (unsigned short*)alloc((size_t)G4H * EE * 2);
    unsigned short* Wihb  = (unsigned short*)alloc((size_t)G4H * EE * 2);
    unsigned short* Whhf  = (unsigned short*)alloc((size_t)G4H * HH * 2);
    unsigned short* Whhb  = (unsigned short*)alloc((size_t)G4H * HH * 2);
    unsigned short* Wout  = (unsigned short*)alloc((size_t)KPAD * EE * 2);
    float*          Biasf = (float*)alloc(G4H * 4);
    float*          Biasb = (float*)alloc(G4H * 4);
    float*          Em    = (float*)alloc((size_t)TT * BB * KK * 4);
    float*          Nll   = (float*)alloc(64 * 4);

    const int CT = 256;
    auto cdiv = [](int a, int b) { return (a + b - 1) / b; };

    // 1) pack everything to bf16 / combine biases
    int nx = BB * TT * EE;
    k_f2bf<<<cdiv(nx, CT), CT, 0, stream>>>(embeds, Xbf, nx);
    k_f2bf<<<cdiv(G4H * EE, CT), CT, 0, stream>>>(w_ih_f, Wihf, G4H * EE);
    k_f2bf<<<cdiv(G4H * EE, CT), CT, 0, stream>>>(w_ih_b, Wihb, G4H * EE);
    k_f2bf<<<cdiv(G4H * HH, CT), CT, 0, stream>>>(w_hh_f, Whhf, G4H * HH);
    k_f2bf<<<cdiv(G4H * HH, CT), CT, 0, stream>>>(w_hh_b, Whhb, G4H * HH);
    k_packwout<<<cdiv(KPAD * EE, CT), CT, 0, stream>>>(w_out, Wout);
    k_bias<<<cdiv(G4H, CT), CT, 0, stream>>>(b_ih_f, b_hh_f, Biasf);
    k_bias<<<cdiv(G4H, CT), CT, 0, stream>>>(b_ih_b, b_hh_b, Biasb);

    dim3 gPre(G4H / 64, (BB * TT) / 64);       // (32, 512): 64x64 tiles
    dim3 gRec(G4H / 64, BB / 16);              // (32, 4): 16x16 tiles
    const size_t pgStride = (size_t)BB * G4H;  // per-timestep slice

    // 2) forward direction
    k_pregates<<<gPre, 128, 0, stream>>>(Xbf, Wihf, Biasf, PG);
    k_init<<<cdiv(BB * HH, CT), CT, 0, stream>>>(h0, c0, Hcur, Ccur);
    for (int t = 0; t < TT; t++) {
        k_recgemm<<<gRec, 128, 0, stream>>>(Hcur, Whhf, PG + (size_t)t * pgStride, Gtmp);
        k_cell<<<cdiv(BB * HH, CT), CT, 0, stream>>>(Gtmp, Ccur, Hcur,
                                                     HFbf + (size_t)t * BB * HH);
    }

    // 3) backward direction (reuses PG buffer; iterates t = T-1 .. 0)
    k_pregates<<<gPre, 128, 0, stream>>>(Xbf, Wihb, Biasb, PG);
    k_init<<<cdiv(BB * HH, CT), CT, 0, stream>>>(h0 + BB * HH, c0 + BB * HH, Hcur, Ccur);
    for (int s = 0; s < TT; s++) {
        int t = TT - 1 - s;
        k_recgemm<<<gRec, 128, 0, stream>>>(Hcur, Whhb, PG + (size_t)t * pgStride, Gtmp);
        k_cell<<<cdiv(BB * HH, CT), CT, 0, stream>>>(Gtmp, Ccur, Hcur,
                                                     HBbf + (size_t)t * BB * HH);
    }

    // 4) emissions
    dim3 gOut(KPAD / 16, (TT * BB) / 16);      // (5, 2048), 1 wave/block
    k_outproj<<<gOut, 32, 0, stream>>>(HFbf, HBbf, Wout, b_out, Em);

    // 5) CRF NLL per sequence + mean
    k_crf<<<64, 128, 0, stream>>>(Em, tag_ids, lengths, start_t, end_t, trans, Nll);
    k_reduce<<<1, 64, 0, stream>>>(Nll, (float*)d_out);
}
